// CausalSelfAttention_86165633892562
// MI455X (gfx1250) — matrix operand order repaired and checked by reading, not compiled, not run on hardware
//
#include <hip/hip_runtime.h>
#include <hip/hip_bf16.h>

// ---------------------------------------------------------------------------
// Causal self-attention (B=2, T=2048, DIM=2048, H=16, KVH=4, HD=128, G=4)
// bf16 WMMA pipeline for MI455X / gfx1250 (wave32).
//   - one-time f32->bf16 conversion of x / Wq / Wkv / Wo
//   - double-buffered bf16 GEMM with async global->LDS staging (ASYNCcnt)
//   - flash attention with causal tile skipping
// ---------------------------------------------------------------------------

#define B_   2
#define T_   2048
#define DIM_ 2048
#define H_   16
#define KVH_ 4
#define HD_  128
#define G_   (H_ / KVH_)

typedef __attribute__((ext_vector_type(16))) __bf16 v16bf;
typedef __attribute__((ext_vector_type(8)))  float  v8f;
typedef __attribute__((ext_vector_type(4)))  int    v4i;

typedef __attribute__((address_space(1))) v4i as1_v4i;
typedef __attribute__((address_space(3))) v4i as3_v4i;

union BF16x16 { v16bf v; unsigned int u[8]; };

static __device__ __forceinline__ unsigned short f2bf(float f) {
    unsigned int u = __float_as_uint(f);
    unsigned int r = u + 0x7FFFu + ((u >> 16) & 1u);   // round-to-nearest-even
    return (unsigned short)(r >> 16);
}
static __device__ __forceinline__ unsigned int pack2(unsigned short lo, unsigned short hi) {
    return (unsigned int)lo | ((unsigned int)hi << 16);
}
static __device__ __forceinline__ v8f wmma_bf16(const BF16x16& a, const BF16x16& b, v8f c) {
    return __builtin_amdgcn_wmma_f32_16x16x32_bf16(false, a.v, false, b.v, (short)0, c, false, false);
}
static __device__ __forceinline__ v8f v8f_zero() {
    v8f z;
    #pragma unroll
    for (int i = 0; i < 8; ++i) z[i] = 0.0f;
    return z;
}

// ---- async global->LDS copy of 16B (8 bf16), with synchronous fallback ----
#if defined(__has_builtin)
#if __has_builtin(__builtin_amdgcn_global_load_async_to_lds_b128)
#define USE_ASYNC_LDS 1
#endif
#endif

static __device__ __forceinline__ void async_cp16(const unsigned short* g, unsigned short* l) {
#ifdef USE_ASYNC_LDS
    // AS1/AS3 pointers built via integer cast: numerically exact on AMDGPU
    // (generic global == AS1 value; generic LDS low 32 bits == LDS offset).
    as1_v4i* gp = (as1_v4i*)(unsigned long long)g;
    as3_v4i* lp = (as3_v4i*)(unsigned int)(unsigned long long)l;
    __builtin_amdgcn_global_load_async_to_lds_b128(gp, lp, 0, 0);
#else
    *(uint4*)l = *(const uint4*)g;
#endif
}
static __device__ __forceinline__ void wait_async_all() {
#ifdef USE_ASYNC_LDS
#if __has_builtin(__builtin_amdgcn_s_wait_asynccnt)
    __builtin_amdgcn_s_wait_asynccnt(0);
#else
    asm volatile("s_wait_asynccnt 0" ::: "memory");
#endif
#endif
}

// ---------------------------------------------------------------------------
// One-time f32 -> bf16 conversion (4 elements / thread).
// ---------------------------------------------------------------------------
__global__ __launch_bounds__(256) void f32_to_bf16_kernel(const float* __restrict__ src,
                                                          unsigned short* __restrict__ dst)
{
    const size_t i4 = ((size_t)blockIdx.x * 256 + threadIdx.x) * 4;
    float4 v = *(const float4*)&src[i4];
    uint2 u; u.x = pack2(f2bf(v.x), f2bf(v.y)); u.y = pack2(f2bf(v.z), f2bf(v.w));
    *(uint2*)&dst[i4] = u;
}

// ---------------------------------------------------------------------------
// C = A * W^T, all-bf16 operands, f32 result.  A: MxK, W: NxK, C: MxN.
// 128x128 block, BK=32, double-buffered LDS with async global->LDS staging.
// 256 threads = 8 waves; wave (wm,wn) owns 32x64 = 2x4 WMMA accumulators.
// ---------------------------------------------------------------------------
__global__ __launch_bounds__(256) void gemm_bf16_kernel(const unsigned short* __restrict__ A,
                                                        const unsigned short* __restrict__ Wb,
                                                        float* __restrict__ C,
                                                        int M, int N, int K)
{
    constexpr int BK  = 32;
    constexpr int LDA = BK + 8;                       // 80B row stride (16B aligned)
    __shared__ unsigned short ldsA[2][128][LDA];
    __shared__ unsigned short ldsB[2][128][LDA];

    const int tid  = threadIdx.x;
    const int lane = tid & 31;
    const int lr   = lane & 15;
    const int lh   = lane >> 4;
    const int w    = tid >> 5;
    const int wm   = w & 3;
    const int wn   = w >> 2;

    const int n0 = blockIdx.x * 128;
    const int m0 = blockIdx.y * 128;

    v8f acc[2][4];
    #pragma unroll
    for (int i = 0; i < 2; ++i)
        #pragma unroll
        for (int j = 0; j < 4; ++j) acc[i][j] = v8f_zero();

    // stage one 128x32 bf16 tile of A and W into LDS buffer `buf`
    auto issue = [&](int kk, int buf) {
        #pragma unroll
        for (int i = 0; i < 2; ++i) {
            const int row = i * 64 + (tid >> 2);
            const int c8  = (tid & 3) * 8;
            async_cp16(&A [(size_t)(m0 + row) * K + kk + c8], &ldsA[buf][row][c8]);
            async_cp16(&Wb[(size_t)(n0 + row) * K + kk + c8], &ldsB[buf][row][c8]);
        }
    };

    const int steps = K / BK;
    issue(0, 0);
    for (int s = 0; s < steps; ++s) {
        const int cur = s & 1;
        wait_async_all();
        __syncthreads();                              // tile `cur` resident in LDS
        if (s + 1 < steps) issue((s + 1) * BK, cur ^ 1);  // overlap next stage with compute

        BF16x16 a[2], b[4];
        #pragma unroll
        for (int i = 0; i < 2; ++i) {
            const int mr = wm * 32 + i * 16;
            #pragma unroll
            for (int q = 0; q < 8; ++q) {
                const int k = ((q >> 2) << 4) + (lh << 3) + ((q & 3) << 1);
                a[i].u[q] = *(const unsigned int*)&ldsA[cur][mr + lr][k];
            }
        }
        #pragma unroll
        for (int j = 0; j < 4; ++j) {
            const int nr = wn * 64 + j * 16;
            #pragma unroll
            for (int q = 0; q < 8; ++q) {
                const int k = ((q >> 2) << 4) + (lh << 3) + ((q & 3) << 1);
                b[j].u[q] = *(const unsigned int*)&ldsB[cur][nr + lr][k];
            }
        }
        #pragma unroll
        for (int i = 0; i < 2; ++i)
            #pragma unroll
            for (int j = 0; j < 4; ++j)
                acc[i][j] = wmma_bf16(a[i], b[j], acc[i][j]);
        __syncthreads();                              // reads of `cur` done
    }

    #pragma unroll
    for (int i = 0; i < 2; ++i)
        #pragma unroll
        for (int j = 0; j < 4; ++j)
            #pragma unroll
            for (int r = 0; r < 8; ++r) {
                const int row = m0 + wm * 32 + i * 16 + lh * 8 + r;
                const int col = n0 + wn * 64 + j * 16 + lr;
                C[(size_t)row * N + col] = acc[i][j][r];
            }
}

// ---------------------------------------------------------------------------
// RMSNorm + RoPE.  One wave per (b,t,head) row of 128.  Output bf16.
// ---------------------------------------------------------------------------
__device__ __forceinline__ void norm_rope_row(const float* __restrict__ src,
                                              unsigned short* __restrict__ dst,
                                              int t, int lane, float extra_scale)
{
    float4 xv = *(const float4*)&src[lane * 4];
    float ss = xv.x * xv.x + xv.y * xv.y + xv.z * xv.z + xv.w * xv.w;
    #pragma unroll
    for (int off = 16; off > 0; off >>= 1) ss += __shfl_xor(ss, off, 32);
    const float r = rsqrtf(ss * (1.0f / 128.0f) + 1.1920929e-7f) * extra_scale;

    float ov[4] = { xv.x * r, xv.y * r, xv.z * r, xv.w * r };
    float pv[4];
    #pragma unroll
    for (int c = 0; c < 4; ++c) pv[c] = __shfl_xor(ov[c], 16, 32);

    const bool low = (lane < 16);
    unsigned short o[4];
    #pragma unroll
    for (int c = 0; c < 4; ++c) {
        const int i = lane * 4 + c;
        const int j = i & 63;                                  // frequency index
        const float ang = (float)t * __expf(-(float)j * 0.14391156646f); // ln(1e4)/64
        float sn, cs; __sincosf(ang, &sn, &cs);
        const float x1 = low ? ov[c] : pv[c];
        const float x2 = low ? pv[c] : ov[c];
        const float out = low ? (x1 * cs + x2 * sn) : (-x1 * sn + x2 * cs);
        o[c] = f2bf(out);
    }
    uint2 u; u.x = pack2(o[0], o[1]); u.y = pack2(o[2], o[3]);
    *(uint2*)&dst[lane * 4] = u;
}

__global__ __launch_bounds__(256) void rope_q_kernel(const float* __restrict__ qf,
                                                     unsigned short* __restrict__ qbf)
{
    const int wid  = blockIdx.x * 8 + (threadIdx.x >> 5);      // (b*T + t)*H + h
    const int lane = threadIdx.x & 31;
    const int h = wid % H_;
    const int t = (wid / H_) % T_;
    const int b = wid / (H_ * T_);
    const float* src = qf + (size_t)wid * HD_;
    unsigned short* dst = qbf + (((size_t)(b * H_ + h) * T_) + t) * HD_;
    norm_rope_row(src, dst, t, lane, 0.08838834764831845f);    // 1/sqrt(128)
}

__global__ __launch_bounds__(256) void rope_kv_kernel(const float* __restrict__ kvf,
                                                      unsigned short* __restrict__ kbf,
                                                      unsigned short* __restrict__ vbf)
{
    const int wid  = blockIdx.x * 8 + (threadIdx.x >> 5);      // (b*T + t)*KVH + kvh
    const int lane = threadIdx.x & 31;
    const int kvh = wid % KVH_;
    const int t   = (wid / KVH_) % T_;
    const int b   = wid / (KVH_ * T_);

    const float* ksrc = kvf + ((((size_t)(b * T_ + t) * 2 + 0) * KVH_) + kvh) * HD_;
    const float* vsrc = kvf + ((((size_t)(b * T_ + t) * 2 + 1) * KVH_) + kvh) * HD_;
    unsigned short* kdst = kbf + (((size_t)(b * KVH_ + kvh) * T_) + t) * HD_;
    unsigned short* vdst = vbf + (((size_t)(b * KVH_ + kvh) * T_) + t) * HD_;

    norm_rope_row(ksrc, kdst, t, lane, 1.0f);

    float4 vv = *(const float4*)&vsrc[lane * 4];
    uint2 u; u.x = pack2(f2bf(vv.x), f2bf(vv.y)); u.y = pack2(f2bf(vv.z), f2bf(vv.w));
    *(uint2*)&vdst[lane * 4] = u;
}

// ---------------------------------------------------------------------------
// Flash attention.  grid = (B*H, T/128).  256 threads = 8 waves; wave w owns a
// 16-row q strip.  64-wide K/V tiles staged in LDS (K via async copy; V
// transposed so P*V B-fragments read K-contiguous pairs).  Causal tiles that
// are fully masked for a wave are skipped (wave-uniform branch).
// ---------------------------------------------------------------------------
__global__ __launch_bounds__(256) void attn_kernel(const unsigned short* __restrict__ qbf,
                                                   const unsigned short* __restrict__ kbf,
                                                   const unsigned short* __restrict__ vbf,
                                                   unsigned short* __restrict__ obf)
{
    __shared__ unsigned short ldsK[64][136];      // [krow][hd] row-major
    __shared__ unsigned short ldsVt[128][72];     // [hd][krow] transposed
    __shared__ unsigned short ldsP[8][16][72];    // per-wave P staging

    const int tid  = threadIdx.x;
    const int lane = tid & 31;
    const int lr   = lane & 15;
    const int lh   = lane >> 4;
    const int w    = tid >> 5;

    const int bh  = blockIdx.x;
    const int b   = bh / H_;
    const int h   = bh % H_;
    const int kvh = h / G_;
    const int qbase = blockIdx.y * 128;
    const int qr0   = qbase + w * 16;

    const unsigned short* qhead = qbf + ((size_t)(b * H_ + h) * T_) * HD_;
    const unsigned short* khead = kbf + ((size_t)(b * KVH_ + kvh) * T_) * HD_;
    const unsigned short* vhead = vbf + ((size_t)(b * KVH_ + kvh) * T_) * HD_;

    // Q fragments stay in registers across the whole k loop (16x128 strip).
    BF16x16 qa[4];
    #pragma unroll
    for (int ks = 0; ks < 4; ++ks)
        #pragma unroll
        for (int q = 0; q < 8; ++q) {
            const int k = ks * 32 + ((q >> 2) << 4) + (lh << 3) + ((q & 3) << 1);
            qa[ks].u[q] = *(const unsigned int*)&qhead[(size_t)(qr0 + lr) * HD_ + k];
        }

    v8f o[8];
    float m_i[8], l_i[8];
    #pragma unroll
    for (int n = 0; n < 8; ++n) o[n] = v8f_zero();
    #pragma unroll
    for (int r = 0; r < 8; ++r) { m_i[r] = -3.0e38f; l_i[r] = 0.0f; }

    const int ktiles = qbase / 64 + 2;            // cover columns 0 .. qbase+127
    for (int kt = 0; kt < ktiles; ++kt) {
        const int kb = kt * 64;
        // ---- stage K (async copy, row-major) and V (manual transpose) ----
        #pragma unroll
        for (int p = 0; p < 4; ++p) {
            const int r = p * 16 + (tid >> 4);
            const int c = (tid & 15) * 8;
            async_cp16(&khead[(size_t)(kb + r) * HD_ + c], &ldsK[r][c]);
            uint4 vv = *(const uint4*)&vhead[(size_t)(kb + r) * HD_ + c];
            const unsigned short* sp = (const unsigned short*)&vv;
            #pragma unroll
            for (int e = 0; e < 8; ++e) ldsVt[c + e][r] = sp[e];
        }
        wait_async_all();
        __syncthreads();

        if (kb <= qr0 + 15) {                     // wave-uniform causal skip
            // ---- S = Q * K^T  (16 x 64 strip, 4 n-tiles x 4 k-steps) ----
            v8f s[4];
            #pragma unroll
            for (int j = 0; j < 4; ++j) {
                v8f a = v8f_zero();
                #pragma unroll
                for (int ks = 0; ks < 4; ++ks) {
                    BF16x16 bk;
                    #pragma unroll
                    for (int q = 0; q < 8; ++q) {
                        const int k = ks * 32 + ((q >> 2) << 4) + (lh << 3) + ((q & 3) << 1);
                        bk.u[q] = *(const unsigned int*)&ldsK[j * 16 + lr][k];
                    }
                    a = wmma_bf16(qa[ks], bk, a);
                }
                s[j] = a;
            }

            // ---- causal mask + online softmax ----
            #pragma unroll
            for (int j = 0; j < 4; ++j) {
                const int c = kb + j * 16 + lr;
                #pragma unroll
                for (int r = 0; r < 8; ++r) {
                    const int qg = qr0 + lh * 8 + r;
                    if (c > qg) s[j][r] = -3.0e38f;
                }
            }
            float mx[8];
            #pragma unroll
            for (int r = 0; r < 8; ++r) {
                mx[r] = fmaxf(fmaxf(s[0][r], s[1][r]), fmaxf(s[2][r], s[3][r]));
                #pragma unroll
                for (int off = 8; off > 0; off >>= 1)
                    mx[r] = fmaxf(mx[r], __shfl_xor(mx[r], off, 32));
            }
            float alpha[8], rs[8];
            #pragma unroll
            for (int r = 0; r < 8; ++r) {
                const float mnew = fmaxf(m_i[r], mx[r]);
                alpha[r] = __expf(m_i[r] - mnew);
                m_i[r] = mnew;
                rs[r] = 0.0f;
            }
            #pragma unroll
            for (int j = 0; j < 4; ++j)
                #pragma unroll
                for (int r = 0; r < 8; ++r) {
                    const float pvv = __expf(s[j][r] - m_i[r]);
                    s[j][r] = pvv;
                    rs[r] += pvv;
                }
            #pragma unroll
            for (int r = 0; r < 8; ++r) {
                #pragma unroll
                for (int off = 8; off > 0; off >>= 1) rs[r] += __shfl_xor(rs[r], off, 32);
                l_i[r] = l_i[r] * alpha[r] + rs[r];
            }
            #pragma unroll
            for (int n = 0; n < 8; ++n)
                #pragma unroll
                for (int r = 0; r < 8; ++r) o[n][r] *= alpha[r];

            // ---- P: D layout -> A layout via per-wave LDS staging ----
            #pragma unroll
            for (int j = 0; j < 4; ++j)
                #pragma unroll
                for (int r = 0; r < 8; ++r)
                    ldsP[w][lh * 8 + r][j * 16 + lr] = f2bf(s[j][r]);

            BF16x16 pa[2];
            #pragma unroll
            for (int ks = 0; ks < 2; ++ks)
                #pragma unroll
                for (int q = 0; q < 8; ++q) {
                    const int k = ks * 32 + ((q >> 2) << 4) + (lh << 3) + ((q & 3) << 1);
                    pa[ks].u[q] = *(const unsigned int*)&ldsP[w][lr][k];
                }

            // ---- O += P * V  (8 n-tiles over HD, 2 k-steps) ----
            #pragma unroll
            for (int n = 0; n < 8; ++n)
                #pragma unroll
                for (int ks = 0; ks < 2; ++ks) {
                    BF16x16 bv;
                    #pragma unroll
                    for (int q = 0; q < 8; ++q) {
                        const int k = ks * 32 + ((q >> 2) << 4) + (lh << 3) + ((q & 3) << 1);
                        bv.u[q] = *(const unsigned int*)&ldsVt[n * 16 + lr][k];
                    }
                    o[n] = wmma_bf16(pa[ks], bv, o[n]);
                }
        }
        __syncthreads();
    }

    // ---- epilogue: normalize by l and store bf16 (B, T, H*HD) ----
    #pragma unroll
    for (int r = 0; r < 8; ++r) l_i[r] = 1.0f / l_i[r];
    #pragma unroll
    for (int n = 0; n < 8; ++n)
        #pragma unroll
        for (int r = 0; r < 8; ++r) {
            const int qg = qr0 + lh * 8 + r;
            obf[((size_t)b * T_ + qg) * (H_ * HD_) + h * HD_ + n * 16 + lr] =
                f2bf(o[n][r] * l_i[r]);
        }
}

// ---------------------------------------------------------------------------
// Host side.  d_ws layout (MB offsets):
//   xbf 0..16, wqbf 16..24, wkvbf 24..28, wobf 28..36,
//   qf 36..68 (f32), kvf 68..84 (f32),
//   qbf 84..100, kbf 100..104, vbf 104..108, obf 108..124   (all bf16)
// ---------------------------------------------------------------------------
extern "C" void kernel_launch(void* const* d_in, const int* in_sizes, int n_in,
                              void* d_out, int out_size, void* d_ws, size_t ws_size,
                              hipStream_t stream)
{
    const float* x   = (const float*)d_in[0];
    const float* Wq  = (const float*)d_in[1];
    const float* Wkv = (const float*)d_in[2];
    const float* Wo  = (const float*)d_in[3];

    char* ws = (char*)d_ws;
    const size_t MB = 1024 * 1024;
    unsigned short* xbf   = (unsigned short*)(ws + 0 * MB);
    unsigned short* wqbf  = (unsigned short*)(ws + 16 * MB);
    unsigned short* wkvbf = (unsigned short*)(ws + 24 * MB);
    unsigned short* wobf  = (unsigned short*)(ws + 28 * MB);
    float*          qf    = (float*)(ws + 36 * MB);
    float*          kvf   = (float*)(ws + 68 * MB);
    unsigned short* qbf   = (unsigned short*)(ws + 84 * MB);
    unsigned short* kbf   = (unsigned short*)(ws + 100 * MB);
    unsigned short* vbf   = (unsigned short*)(ws + 104 * MB);
    unsigned short* obf   = (unsigned short*)(ws + 108 * MB);

    const int M = B_ * T_;   // 4096

    // 0) one-time bf16 conversions (n / 1024 blocks, 4 elems/thread)
    f32_to_bf16_kernel<<<(B_ * T_ * DIM_) / 1024, 256, 0, stream>>>(x, xbf);
    f32_to_bf16_kernel<<<(H_ * HD_ * DIM_) / 1024, 256, 0, stream>>>(Wq, wqbf);
    f32_to_bf16_kernel<<<(2 * KVH_ * HD_ * DIM_) / 1024, 256, 0, stream>>>(Wkv, wkvbf);
    f32_to_bf16_kernel<<<(DIM_ * H_ * HD_) / 1024, 256, 0, stream>>>(Wo, wobf);

    // 1) Q = x @ Wq^T ; KV = x @ Wkv^T
    gemm_bf16_kernel<<<dim3((H_ * HD_) / 128, M / 128), 256, 0, stream>>>(
        xbf, wqbf, qf, M, H_ * HD_, DIM_);
    gemm_bf16_kernel<<<dim3((2 * KVH_ * HD_) / 128, M / 128), 256, 0, stream>>>(
        xbf, wkvbf, kvf, M, 2 * KVH_ * HD_, DIM_);

    // 2) RMSNorm + RoPE (+ bf16 repack)
    rope_q_kernel<<<(B_ * T_ * H_) / 8, 256, 0, stream>>>(qf, qbf);
    rope_kv_kernel<<<(B_ * T_ * KVH_) / 8, 256, 0, stream>>>(kvf, kbf, vbf);

    // 3) Causal flash attention
    attn_kernel<<<dim3(B_ * H_, T_ / 128), 256, 0, stream>>>(qbf, kbf, vbf, obf);

    // 4) out = attn_out @ Wo^T  (f32 result)
    gemm_bf16_kernel<<<dim3(DIM_ / 128, M / 128), 256, 0, stream>>>(
        obf, wobf, (float*)d_out, M, DIM_, H_ * HD_);
}
